// Model_76338748719721
// MI455X (gfx1250) — matrix operand order repaired and checked by reading, not compiled, not run on hardware
//
#include <hip/hip_runtime.h>
#include <stdint.h>

#define D 1024
#define NNODES 65536
#define NEDGES 131072
#define NGRAPH 64
#define EPSBN 1e-5f

typedef __attribute__((ext_vector_type(16))) __bf16 bf16x16;
typedef __attribute__((ext_vector_type(8)))  float  v8f;

// ---------------------------------------------------------------------------
// Weight prep: fold BN scale into weights, build
//   W0eff[k][n] = W0[k][n]*s0[n]                        (f32, tiny GEMM)
//   WA1 [k][n] = (W1top - W1bot)[k][n]*s1[n]            (f32, tiny GEMM)
//   WA2 [k][n] = (W2top - W2bot)[k][n]*s2[n]            (f32)
//   WBt [n(0..2047)][k] = Wbot*s  bf16 hi/lo, COLUMN-major so the WMMA GEMM
//   stages B with contiguous 16B loads (no LDS scatter in the hot loop).
// ---------------------------------------------------------------------------
__global__ void k_weight_prep(const float* __restrict__ W0, const float* __restrict__ g0,
                              const float* __restrict__ W1, const float* __restrict__ g1,
                              const float* __restrict__ W2, const float* __restrict__ g2,
                              float* __restrict__ W0eff, float* __restrict__ WA1,
                              float* __restrict__ WA2,
                              __bf16* __restrict__ WBhi_t, __bf16* __restrict__ WBlo_t) {
    int idx = blockIdx.x * blockDim.x + threadIdx.x;   // 0 .. 1024*1024
    int k = idx >> 10, n = idx & 1023;
    float inv = rsqrtf(1.0f + EPSBN);
    float s0 = g0[n] * inv, s1 = g1[n] * inv, s2 = g2[n] * inv;
    W0eff[idx] = W0[idx] * s0;
    float w1t = W1[(size_t)k * D + n], w1b = W1[(size_t)(D + k) * D + n];
    float w2t = W2[(size_t)k * D + n], w2b = W2[(size_t)(D + k) * D + n];
    WA1[idx] = (w1t - w1b) * s1;
    WA2[idx] = (w2t - w2b) * s2;
    float wb1 = w1b * s1, wb2 = w2b * s2;
    __bf16 h1 = (__bf16)wb1;
    WBhi_t[(size_t)n * D + k] = h1;
    WBlo_t[(size_t)n * D + k] = (__bf16)(wb1 - (float)h1);
    __bf16 h2 = (__bf16)wb2;
    WBhi_t[(size_t)(1024 + n) * D + k] = h2;
    WBlo_t[(size_t)(1024 + n) * D + k] = (__bf16)(wb2 - (float)h2);
}

// Split fp32 x into bf16 hi + bf16 lo (x ~= hi + lo, ~16 mantissa bits)
__global__ void k_split(const float* __restrict__ x, __bf16* __restrict__ xhi,
                        __bf16* __restrict__ xlo, int total) {
    int i = blockIdx.x * blockDim.x + threadIdx.x;
    if (i < total) {
        float v = x[i];
        __bf16 h = (__bf16)v;
        xhi[i] = h;
        xlo[i] = (__bf16)(v - (float)h);
    }
}

__global__ void k_fill_neginf(float* __restrict__ p, int total4) {
    int i = blockIdx.x * blockDim.x + threadIdx.x;
    if (i < total4) {
        float4 v = make_float4(-__builtin_inff(), -__builtin_inff(),
                               -__builtin_inff(), -__builtin_inff());
        ((float4*)p)[i] = v;
    }
}

__global__ void k_degree(const long long* __restrict__ ei, int* __restrict__ indeg) {
    int e = blockIdx.x * blockDim.x + threadIdx.x;
    if (e < NEDGES) atomicAdd(&indeg[(int)ei[NEDGES + e]], 1);
}

// Per-graph sums of x (all nodes) and of x over empty (indeg==0) nodes,
// plus node counts and nonempty counts.
__global__ void k_graph_sums(const float* __restrict__ x, const long long* __restrict__ batch,
                             const int* __restrict__ indeg,
                             float* __restrict__ Ssum, float* __restrict__ Esum,
                             float* __restrict__ cnt, float* __restrict__ nne) {
    int i = blockIdx.x;
    int g = (int)batch[i];
    bool empty = (indeg[i] == 0);
    const float* xr = x + (size_t)i * D;
    float* Sg = Ssum + (size_t)g * D;
    float* Eg = Esum + (size_t)g * D;
    for (int c = threadIdx.x; c < D; c += blockDim.x) {
        float v = xr[c];
        atomicAdd(&Sg[c], v);
        if (empty) atomicAdd(&Eg[c], v);
    }
    if (threadIdx.x == 0) {
        atomicAdd(&cnt[g], 1.0f);
        if (!empty) atomicAdd(&nne[g], 1.0f);
    }
}

// ---------------------------------------------------------------------------
// WMMA GEMM: Y[M x Nc] = (Ahi+Alo)[M x K] @ (Bhi+Blo)[K x Nc], f32 accum.
// B supplied COLUMN-major [Nc][K] so both A and B stage with 16B vector copies.
// Block = 256 threads = 8 waves; block tile 128x128; K step 32.
// Register double-buffering: next K-tile global loads overlap current WMMAs.
// Each wave: one 16-row stripe x 8 N-subtiles x 3 hi/lo WMMAs per K step.
// ---------------------------------------------------------------------------
#define BM 128
#define BN 128
#define BK 32
#define LDSS 40   // padded LDS row stride (elems): 80B rows keep 16B alignment

__global__ __launch_bounds__(256)
void k_gemm_wmma(const __bf16* __restrict__ Ahi, const __bf16* __restrict__ Alo,
                 const __bf16* __restrict__ Bhi_t, const __bf16* __restrict__ Blo_t,
                 float* __restrict__ Cm, int K, int Nc) {
    __shared__ __align__(16) __bf16 sAhi[BM][LDSS];
    __shared__ __align__(16) __bf16 sAlo[BM][LDSS];
    __shared__ __align__(16) __bf16 sBhi[BN][LDSS];   // [n][k]
    __shared__ __align__(16) __bf16 sBlo[BN][LDSS];

    const int tid  = threadIdx.x;
    const int lane = tid & 31;
    const int wave = tid >> 5;
    const int m    = lane & 15;   // M row within subtile (A) / N col (B,C)
    const int h    = lane >> 4;   // lane-half
    const int rowBase = blockIdx.y * BM;
    const int colBase = blockIdx.x * BN;

    v8f acc[8] = {};

    // staging map: each thread owns one 32-elem row half (16B x2) of A and B
    const int sRow  = tid >> 1;        // 0..127
    const int sColH = (tid & 1) * 16;  // 0 or 16

    const __bf16* gAh = Ahi   + (size_t)(rowBase + sRow) * K + sColH;
    const __bf16* gAl = Alo   + (size_t)(rowBase + sRow) * K + sColH;
    const __bf16* gBh = Bhi_t + (size_t)(colBase + sRow) * K + sColH;
    const __bf16* gBl = Blo_t + (size_t)(colBase + sRow) * K + sColH;

    // prologue: fetch K-tile 0 into registers
    uint4 rAh0 = ((const uint4*)gAh)[0], rAh1 = ((const uint4*)gAh)[1];
    uint4 rAl0 = ((const uint4*)gAl)[0], rAl1 = ((const uint4*)gAl)[1];
    uint4 rBh0 = ((const uint4*)gBh)[0], rBh1 = ((const uint4*)gBh)[1];
    uint4 rBl0 = ((const uint4*)gBl)[0], rBl1 = ((const uint4*)gBl)[1];

    for (int k0 = 0; k0 < K; k0 += BK) {
        // commit staged registers to LDS
        *(uint4*)&sAhi[sRow][sColH]     = rAh0;
        *(uint4*)&sAhi[sRow][sColH + 8] = rAh1;
        *(uint4*)&sAlo[sRow][sColH]     = rAl0;
        *(uint4*)&sAlo[sRow][sColH + 8] = rAl1;
        *(uint4*)&sBhi[sRow][sColH]     = rBh0;
        *(uint4*)&sBhi[sRow][sColH + 8] = rBh1;
        *(uint4*)&sBlo[sRow][sColH]     = rBl0;
        *(uint4*)&sBlo[sRow][sColH + 8] = rBl1;
        __syncthreads();

        // prefetch next K-tile from global while WMMAs consume this one
        if (k0 + BK < K) {
            const __bf16* pAh = gAh + (k0 + BK);
            const __bf16* pAl = gAl + (k0 + BK);
            const __bf16* pBh = gBh + (k0 + BK);
            const __bf16* pBl = gBl + (k0 + BK);
            rAh0 = ((const uint4*)pAh)[0]; rAh1 = ((const uint4*)pAh)[1];
            rAl0 = ((const uint4*)pAl)[0]; rAl1 = ((const uint4*)pAl)[1];
            rBh0 = ((const uint4*)pBh)[0]; rBh1 = ((const uint4*)pBh)[1];
            rBl0 = ((const uint4*)pBl)[0]; rBl1 = ((const uint4*)pBl)[1];
        }

        // A fragment (16x32 bf16): lane half h -> K = h*16 + (0..15) contiguous
        bf16x16 aH, aL;
        ((uint4*)&aH)[0] = *(const uint4*)&sAhi[wave * 16 + m][h * 16];
        ((uint4*)&aH)[1] = *(const uint4*)&sAhi[wave * 16 + m][h * 16 + 8];
        ((uint4*)&aL)[0] = *(const uint4*)&sAlo[wave * 16 + m][h * 16];
        ((uint4*)&aL)[1] = *(const uint4*)&sAlo[wave * 16 + m][h * 16 + 8];

#pragma unroll
        for (int nt = 0; nt < 8; ++nt) {
            // B fragment (32x16 bf16): lane n = m, K = h*16 + (0..15) contiguous
            bf16x16 bH, bL;
            ((uint4*)&bH)[0] = *(const uint4*)&sBhi[nt * 16 + m][h * 16];
            ((uint4*)&bH)[1] = *(const uint4*)&sBhi[nt * 16 + m][h * 16 + 8];
            ((uint4*)&bL)[0] = *(const uint4*)&sBlo[nt * 16 + m][h * 16];
            ((uint4*)&bL)[1] = *(const uint4*)&sBlo[nt * 16 + m][h * 16 + 8];
            acc[nt] = __builtin_amdgcn_wmma_f32_16x16x32_bf16(
                false, aH, false, bH, (short)0, acc[nt], false, false);
            acc[nt] = __builtin_amdgcn_wmma_f32_16x16x32_bf16(
                false, aL, false, bH, (short)0, acc[nt], false, false);
            acc[nt] = __builtin_amdgcn_wmma_f32_16x16x32_bf16(
                false, aH, false, bL, (short)0, acc[nt], false, false);
        }
        __syncthreads();
    }

    // C/D f32 layout: lane n = lane%16, vgpr r -> row = h*8 + r
#pragma unroll
    for (int nt = 0; nt < 8; ++nt) {
#pragma unroll
        for (int r = 0; r < 8; ++r) {
            int row = rowBase + wave * 16 + h * 8 + r;
            int col = colBase + nt * 16 + m;
            Cm[(size_t)row * Nc + col] = acc[nt][r];
        }
    }
}

// float atomic max via sign-split int atomics (works with -inf init, no NaNs)
__device__ __forceinline__ void atomicMaxF(float* addr, float val) {
    if (val >= 0.0f) atomicMax((int*)addr, __float_as_int(val));
    else             atomicMin((unsigned int*)addr, __float_as_uint(val));
}

// segment max over edges: M[dst] = max(M[dst], Y[src]) for both 1024-col halves
__global__ void k_segmax(const long long* __restrict__ ei, const float* __restrict__ Y,
                         float* __restrict__ Mx) {
    int e = blockIdx.x;
    int s = (int)ei[e];
    int d = (int)ei[NEDGES + e];
    const float* ys = Y + (size_t)s * 2048;
    float* md = Mx + (size_t)d * 2048;
    for (int c = threadIdx.x; c < 2048; c += blockDim.x)
        atomicMaxF(&md[c], ys[c]);
}

// per-graph sum of segment-max rows over nonempty nodes
__global__ void k_msum(const float* __restrict__ Mx, const long long* __restrict__ batch,
                       const int* __restrict__ indeg, float* __restrict__ Msum) {
    int i = blockIdx.x;
    if (indeg[i] == 0) return;
    int g = (int)batch[i];
    const float* mi = Mx + (size_t)i * 2048;
    float* mg = Msum + (size_t)g * 2048;
    for (int c = threadIdx.x; c < 2048; c += blockDim.x)
        atomicAdd(&mg[c], mi[c]);
}

// tiny f32 GEMM: C[64][1024] = (A - Asub) @ B[1024][1024]; Asub may be null
__global__ void k_small_gemm(const float* __restrict__ A, const float* __restrict__ Asub,
                             const float* __restrict__ B, float* __restrict__ Cc) {
    __shared__ float sA[D];
    int r = blockIdx.y;
    int c = blockIdx.x * 256 + threadIdx.x;
    for (int k = threadIdx.x; k < D; k += 256) {
        float a = A[(size_t)r * D + k];
        if (Asub) a -= Asub[(size_t)r * D + k];
        sA[k] = a;
    }
    __syncthreads();
    float acc = 0.f;
    for (int k = 0; k < D; ++k)
        acc = fmaf(sA[k], B[(size_t)k * D + c], acc);
    Cc[(size_t)r * D + c] = acc;
}

// final combine: acc[g] = p0+p1+p2, out = acc @ Wr + br
__global__ void k_final(const float* __restrict__ P0, const float* __restrict__ T1,
                        const float* __restrict__ T2, const float* __restrict__ Msum,
                        const float* __restrict__ cnt, const float* __restrict__ nne,
                        const float* __restrict__ b0, const float* __restrict__ g0,
                        const float* __restrict__ be0,
                        const float* __restrict__ b1, const float* __restrict__ g1,
                        const float* __restrict__ be1,
                        const float* __restrict__ b2, const float* __restrict__ g2,
                        const float* __restrict__ be2,
                        const float* __restrict__ Wr, const float* __restrict__ br,
                        float* __restrict__ out) {
    __shared__ float red0[256], red1[256];
    int g = blockIdx.x;
    float cdiv = fmaxf(cnt[g], 1.0f);
    float nn = nne[g];
    float inv = rsqrtf(1.0f + EPSBN);
    float d0 = 0.f, d1 = 0.f;
    for (int c = threadIdx.x; c < D; c += 256) {
        float s0 = g0[c] * inv, s1 = g1[c] * inv, s2 = g2[c] * inv;
        float c0 = b0[c] * s0 + be0[c];
        float bias1 = b1[c] * s1 + be1[c];
        float bias2 = b2[c] * s2 + be2[c];
        float p0 = P0[(size_t)g * D + c] / cdiv + c0;
        float p1 = (T1[(size_t)g * D + c] + nn * bias1 + Msum[(size_t)g * 2048 + c]) / cdiv;
        float p2 = (T2[(size_t)g * D + c] + nn * bias2 + Msum[(size_t)g * 2048 + 1024 + c]) / cdiv;
        float a = p0 + p1 + p2;
        d0 = fmaf(a, Wr[c * 2 + 0], d0);
        d1 = fmaf(a, Wr[c * 2 + 1], d1);
    }
    red0[threadIdx.x] = d0; red1[threadIdx.x] = d1;
    __syncthreads();
    for (int s = 128; s > 0; s >>= 1) {
        if (threadIdx.x < s) {
            red0[threadIdx.x] += red0[threadIdx.x + s];
            red1[threadIdx.x] += red1[threadIdx.x + s];
        }
        __syncthreads();
    }
    if (threadIdx.x == 0) {
        out[g * 2 + 0] = red0[0] + br[0];
        out[g * 2 + 1] = red1[0] + br[1];
    }
}

extern "C" void kernel_launch(void* const* d_in, const int* in_sizes, int n_in,
                              void* d_out, int out_size, void* d_ws, size_t ws_size,
                              hipStream_t stream) {
    const float*     x     = (const float*)d_in[0];
    const long long* ei    = (const long long*)d_in[1];
    const long long* batch = (const long long*)d_in[2];
    const float* W0  = (const float*)d_in[3];
    const float* b0  = (const float*)d_in[4];
    const float* g0  = (const float*)d_in[5];
    const float* be0 = (const float*)d_in[6];
    const float* W1  = (const float*)d_in[7];
    const float* b1  = (const float*)d_in[8];
    const float* g1  = (const float*)d_in[9];
    const float* be1 = (const float*)d_in[10];
    const float* W2  = (const float*)d_in[11];
    const float* b2  = (const float*)d_in[12];
    const float* g2  = (const float*)d_in[13];
    const float* be2 = (const float*)d_in[14];
    const float* Wr  = (const float*)d_in[15];
    const float* br  = (const float*)d_in[16];
    float* out = (float*)d_out;

    char* ws = (char*)d_ws;
    size_t off = 0;
    auto alloc = [&](size_t bytes) -> char* {
        char* p = ws + off;
        off += (bytes + 255) & ~(size_t)255;
        return p;
    };
    __bf16* xhi    = (__bf16*)alloc((size_t)NNODES * D * 2);
    __bf16* xlo    = (__bf16*)alloc((size_t)NNODES * D * 2);
    __bf16* WBhi_t = (__bf16*)alloc((size_t)D * 2048 * 2);
    __bf16* WBlo_t = (__bf16*)alloc((size_t)D * 2048 * 2);
    float*  W0eff  = (float*)alloc((size_t)D * D * 4);
    float*  WA1    = (float*)alloc((size_t)D * D * 4);
    float*  WA2    = (float*)alloc((size_t)D * D * 4);
    float*  Y      = (float*)alloc((size_t)NNODES * 2048 * 4);
    float*  Mx     = (float*)alloc((size_t)NNODES * 2048 * 4);
    float*  Ssum   = (float*)alloc((size_t)NGRAPH * D * 4);
    float*  Esum   = (float*)alloc((size_t)NGRAPH * D * 4);
    float*  Msum   = (float*)alloc((size_t)NGRAPH * 2048 * 4);
    float*  P0     = (float*)alloc((size_t)NGRAPH * D * 4);
    float*  T1     = (float*)alloc((size_t)NGRAPH * D * 4);
    float*  T2     = (float*)alloc((size_t)NGRAPH * D * 4);
    float*  cnt    = (float*)alloc(NGRAPH * 4);
    float*  nne    = (float*)alloc(NGRAPH * 4);
    int*    indeg  = (int*)alloc((size_t)NNODES * 4);

    hipMemsetAsync(Ssum, 0, (size_t)NGRAPH * D * 4, stream);
    hipMemsetAsync(Esum, 0, (size_t)NGRAPH * D * 4, stream);
    hipMemsetAsync(Msum, 0, (size_t)NGRAPH * 2048 * 4, stream);
    hipMemsetAsync(cnt, 0, NGRAPH * 4, stream);
    hipMemsetAsync(nne, 0, NGRAPH * 4, stream);
    hipMemsetAsync(indeg, 0, (size_t)NNODES * 4, stream);

    k_fill_neginf<<<(NNODES * 2048 / 4 + 255) / 256, 256, 0, stream>>>(
        Mx, NNODES * 2048 / 4);
    k_weight_prep<<<(D * D + 255) / 256, 256, 0, stream>>>(
        W0, g0, W1, g1, W2, g2, W0eff, WA1, WA2, WBhi_t, WBlo_t);
    k_split<<<(NNODES * D + 255) / 256, 256, 0, stream>>>(x, xhi, xlo, NNODES * D);
    k_degree<<<(NEDGES + 255) / 256, 256, 0, stream>>>(ei, indeg);
    k_graph_sums<<<NNODES, 256, 0, stream>>>(x, batch, indeg, Ssum, Esum, cnt, nne);

    dim3 gemmGrid(2048 / BN, NNODES / BM);
    k_gemm_wmma<<<gemmGrid, 256, 0, stream>>>(xhi, xlo, WBhi_t, WBlo_t, Y, D, 2048);

    k_segmax<<<NEDGES, 256, 0, stream>>>(ei, Y, Mx);
    k_msum<<<NNODES, 256, 0, stream>>>(Mx, batch, indeg, Msum);

    dim3 sg(4, NGRAPH);
    k_small_gemm<<<sg, 256, 0, stream>>>(Ssum, nullptr, W0eff, P0);
    k_small_gemm<<<sg, 256, 0, stream>>>(Ssum, Esum, WA1, T1);
    k_small_gemm<<<sg, 256, 0, stream>>>(Ssum, Esum, WA2, T2);

    k_final<<<NGRAPH, 256, 0, stream>>>(P0, T1, T2, Msum, cnt, nne,
                                        b0, g0, be0, b1, g1, be1, b2, g2, be2,
                                        Wr, br, out);
}
